// EnergyPredictor_V3_Huang_88081189306545
// MI455X (gfx1250) — matrix operand order repaired and checked by reading, not compiled, not run on hardware
//
#include <hip/hip_runtime.h>
#include <hip/hip_bf16.h>

#define EPSF   1e-8f
#define RCUT   4.8f
#define NBATCH 32768
#define NNBR   64
#define NSEL   20
#define BPB    4               // batch elements per block
#define ROWS   (BPB * NSEL)    // 80 MLP rows per block
#define NWAVE  5               // 5 wave32 = 5 row tiles of 16
#define TPB    (NWAVE * 32)    // 160 threads
#define NFRAG  22              // pre-swizzled weight B-fragments

typedef __attribute__((ext_vector_type(16))) _Float16 v16h;
typedef __attribute__((ext_vector_type(8)))  float    v8f;

// A-fragment (16x32 f16, wave32) element->K map: contiguous K = 16*hh + e (matches fragW B staging)
__device__ __forceinline__ int akmap(int e, int hh) {
  return (hh << 4) + e;
}

__device__ __forceinline__ v8f wmma16(v16h a, v16h b, v8f c) {
  return __builtin_amdgcn_wmma_f32_16x16x32_f16(
      /*neg_a=*/false, a, /*neg_b=*/false, b,
      /*c_mod=*/(short)0, c, /*reuse_a=*/false, /*reuse_b=*/false);
}

// CDNA5 native tanh (TRANS class); avoids branchy libm expansion.
__device__ __forceinline__ float fast_tanh(float x) {
#if __has_builtin(__builtin_amdgcn_tanhf)
  return __builtin_amdgcn_tanhf(x);
#elif __has_builtin(__builtin_amdgcn_tanh_f32)
  return __builtin_amdgcn_tanh_f32(x);
#else
  float y;
  asm("v_tanh_f32 %0, %1\n\tv_nop\n\tv_nop" : "=v"(y) : "v"(x));
  return y;
#endif
}

// b^p for b>0 via native v_log_f32 / v_exp_f32
__device__ __forceinline__ float fast_pow_pos(float b, float p) {
  return __builtin_amdgcn_exp2f(p * __builtin_amdgcn_logf(b));
}

__global__ void __launch_bounds__(TPB)
energy_force_torque_fused(const float* __restrict__ dr,
                          const float* __restrict__ ori,
                          const float* __restrict__ nori,
                          const float* __restrict__ W0g,
                          const float* __restrict__ b0g,
                          const float* __restrict__ W1g,
                          const float* __restrict__ b1g,
                          const float* __restrict__ W2g,
                          const float* __restrict__ b2g,
                          const float* __restrict__ f1g,
                          const float* __restrict__ f2g,
                          float* __restrict__ out)
{
  __shared__ float    Rsh[BPB * NNBR];
  __shared__ float    Osh[BPB * 9];
  __shared__ int      selIdx[BPB * NSEL];
  __shared__ float    newRsh[ROWS];
  __shared__ float    fcRsh[ROWS];
  __shared__ float    priorSh[ROWS];
  __shared__ float    drnSel[ROWS * 3];
  __shared__ float    NOsel[ROWS * 9];
  __shared__ _Float16 featSh[ROWS * 16];
  __shared__ float    gfeatSh[ROWS * 16];
  // All weight B-fragments pre-swizzled to per-lane order:
  // fragW[(frag*32 + lane)*16 + e]; each fragment load = one contiguous v16h.
  // frag 0..3   : L0 fwd  W0 (K=32 pad from 15), n-tile t=frag
  // frag 4..11  : L1 fwd  W1, t=(f-4)>>1, Kchunk q=(f-4)&1
  // frag 12..19 : L1 bwd  W1^T, t=(f-12)>>1, q=(f-12)&1
  // frag 20..21 : L0 bwd  W0^T (16 cols, pad from 15), q=f-20
  __shared__ __align__(32) _Float16 fragW[NFRAG * 32 * 16];
  __shared__ float    b0sh[64], b1sh[64], W2sh[64];
  __shared__ float    b2f, c1s[3], pjs[3];
  __shared__ float    scratch[NWAVE * 16 * 64]; // per-wave C<->A re-stage tile
  __shared__ float    Facc[BPB], Sacc[BPB], gOAcc[BPB * 9];

  const int tid = threadIdx.x;
  const int bb  = blockIdx.x * BPB;

  // ------- Phase 0: stage weights f32->f16 in fragment order; scalars -----
  for (int i = tid; i < NFRAG * 512; i += TPB) {
    int frag = i >> 9;
    int l    = (i >> 4) & 31;
    int e    = i & 15;
    int mm   = l & 15, hh2 = l >> 4;
    float v;
    if (frag < 4) {                    // L0 forward
      int t = frag, k = hh2 * 16 + e;
      v = (k < 15) ? W0g[k * 64 + t * 16 + mm] : 0.f;
    } else if (frag < 12) {            // L1 forward
      int t = (frag - 4) >> 1, q = (frag - 4) & 1;
      int k = q * 32 + hh2 * 16 + e;
      v = W1g[k * 64 + t * 16 + mm];
    } else if (frag < 20) {            // L1 backward (W1^T)
      int t = (frag - 12) >> 1, q = (frag - 12) & 1;
      int k = q * 32 + hh2 * 16 + e;
      v = W1g[(t * 16 + mm) * 64 + k];
    } else {                           // L0 backward (W0^T)
      int q = frag - 20;
      int k = q * 32 + hh2 * 16 + e;
      v = (mm < 15) ? W0g[mm * 64 + k] : 0.f;
    }
    fragW[i] = (_Float16)v;
  }
  if (tid < 64) { b0sh[tid] = b0g[tid]; b1sh[tid] = b1g[tid]; W2sh[tid] = W2g[tid]; }
  if (tid == 0) {
    b2f = b2g[0];
    for (int j = 0; j < 3; ++j) {
      c1s[j] = f1g[j] * f1g[j] + EPSF;
      pjs[j] = f2g[j] * f2g[j] + EPSF;
    }
  }

  // ------- Phase 1: distances (dr + EPS) ----------------------------------
  for (int t = tid; t < BPB * NNBR; t += TPB) {
    int bl = t >> 6, i = t & 63;
    long base = ((long)(bb + bl) * NNBR + i) * 3;
    float x = dr[base + 0] + EPSF, y = dr[base + 1] + EPSF, z = dr[base + 2] + EPSF;
    Rsh[t] = sqrtf(x * x + y * y + z * z);
  }
  for (int t = tid; t < BPB * 9; t += TPB) Osh[t] = ori[(long)bb * 9 + t];
  __syncthreads();

  // ------- Phase 2: stable top-20 selection by rank -----------------------
  for (int t = tid; t < BPB * NNBR; t += TPB) {
    int bl = t >> 6, i = t & 63;
    float Ri = Rsh[t];
    int rank = 0;
    for (int j = 0; j < NNBR; ++j) {
      float Rj = Rsh[bl * NNBR + j];
      rank += (Rj < Ri) || (Rj == Ri && j < i);   // matches stable argsort
    }
    if (rank < NSEL) { selIdx[bl * NSEL + rank] = i; newRsh[bl * NSEL + rank] = Ri; }
  }
  __syncthreads();

  // ------- Phase 3: gather + features + cutoff ----------------------------
  if (tid < ROWS) {
    int bl = tid / NSEL;
    int i  = selIdx[tid];
    int g  = bb + bl;
    float R = newRsh[tid];
    long dbase = ((long)g * NNBR + i) * 3;
    float dnv[3];
    dnv[0] = (dr[dbase + 0] + EPSF) / R;
    dnv[1] = (dr[dbase + 1] + EPSF) / R;
    dnv[2] = (dr[dbase + 2] + EPSF) / R;
    drnSel[tid * 3 + 0] = dnv[0]; drnSel[tid * 3 + 1] = dnv[1]; drnSel[tid * 3 + 2] = dnv[2];
    long nbase = ((long)g * NNBR + i) * 9;
    float NO[9];
#pragma unroll
    for (int q = 0; q < 9; ++q) { NO[q] = nori[nbase + q]; NOsel[tid * 9 + q] = NO[q]; }
    const float* O = &Osh[bl * 9];
    float ft[15];
#pragma unroll
    for (int h = 0; h < 3; ++h) {
      float s0 = 0.f, s1 = 0.f;
#pragma unroll
      for (int k = 0; k < 3; ++k) { s0 += dnv[k] * O[k * 3 + h]; s1 += dnv[k] * NO[k * 3 + h]; }
      ft[h] = s0; ft[3 + h] = s1;
    }
#pragma unroll
    for (int l = 0; l < 3; ++l)
#pragma unroll
      for (int mm = 0; mm < 3; ++mm) {
        float s = 0.f;
#pragma unroll
        for (int h = 0; h < 3; ++h) s += O[h * 3 + l] * NO[h * 3 + mm];
        ft[6 + l * 3 + mm] = s;
      }
#pragma unroll
    for (int c = 0; c < 15; ++c) featSh[tid * 16 + c] = (_Float16)ft[c];
    featSh[tid * 16 + 15] = (_Float16)0.f;
    fcRsh[tid] = (R > RCUT) ? 0.f
               : (0.5f * __cosf(3.14159265358979f * R / RCUT) + 0.5f);
  }
  __syncthreads();

  // Phase 3b: F_b = sum_n fcR (deterministic reduction)
  if (tid < BPB) {
    float s = 0.f;
    for (int n = 0; n < NSEL; ++n) s += fcRsh[tid * NSEL + n];
    Facc[tid] = s;
  }
  __syncthreads();

  // ------- Phase 4: WMMA MLP forward + analytic backward ------------------
  const int w    = tid >> 5;
  const int lane = tid & 31;
  const int m    = lane & 15;
  const int hh   = lane >> 4;
  const int row0 = w * 16;
  float* scr = &scratch[w * 16 * 64];
  const v16h* fw = (const v16h*)fragW;

  // feature A-fragment (16 rows x 32 K, cols >=16 are zero pad)
  v16h aF;
#pragma unroll
  for (int e = 0; e < 16; ++e) {
    int K = akmap(e, hh);
    aF[e] = (K < 16) ? featSh[(row0 + m) * 16 + K] : (_Float16)0.f;
  }
  // Layer 0: [16,32] x [32,64] -> tanh -> h1 (4 WMMAs)
#pragma unroll
  for (int t = 0; t < 4; ++t) {
    v16h bW = fw[t * 32 + lane];
    v8f c = {};
    c = wmma16(aF, bW, c);
#pragma unroll
    for (int r = 0; r < 8; ++r)
      scr[(r + 8 * hh) * 64 + t * 16 + m] = fast_tanh(c[r] + b0sh[t * 16 + m]);
  }
  __syncthreads();
  v16h h1a, h1b;
#pragma unroll
  for (int e = 0; e < 16; ++e) {
    int K = akmap(e, hh);
    h1a[e] = (_Float16)scr[m * 64 + K];
    h1b[e] = (_Float16)scr[m * 64 + 32 + K];
  }
  __syncthreads();
  // Layer 1: [16,64] x [64,64] -> tanh -> h2 (8 WMMAs)
#pragma unroll
  for (int t = 0; t < 4; ++t) {
    v16h bA = fw[(4 + t * 2 + 0) * 32 + lane];
    v16h bB = fw[(4 + t * 2 + 1) * 32 + lane];
    v8f c = {};
    c = wmma16(h1a, bA, c);
    c = wmma16(h1b, bB, c);
#pragma unroll
    for (int r = 0; r < 8; ++r)
      scr[(r + 8 * hh) * 64 + t * 16 + m] = fast_tanh(c[r] + b1sh[t * 16 + m]);
  }
  __syncthreads();
  v16h h2a, h2b;
#pragma unroll
  for (int e = 0; e < 16; ++e) {
    int K = akmap(e, hh);
    h2a[e] = (_Float16)scr[m * 64 + K];
    h2b[e] = (_Float16)scr[m * 64 + 32 + K];
  }

  // Layer 2 (N=1): VALU dot + lane16 reduce; then prior + head gradient
  float part = 0.f;
#pragma unroll
  for (int e = 0; e < 16; ++e) {
    int K = akmap(e, hh);
    part += (float)h2a[e] * W2sh[K] + (float)h2b[e] * W2sh[32 + K];
  }
  part += __shfl_xor(part, 16, 32);
  float y3  = part + b2f;
  float enc = fast_tanh(y3);
  float ec  = enc * enc + EPSF;
  int   row = row0 + m;
  int   bl4 = row / NSEL;
  float Rn  = newRsh[row];
  float prior = 0.f, dpe = 0.f;
#pragma unroll
  for (int j = 0; j < 3; ++j) {
    float bj  = c1s[j] * (Rn - ec);                 // > 0 by construction
    float pj  = pjs[j];
    float t0  = fast_pow_pos(bj, -pj);              // bj^(-pj)
    prior += t0;
    dpe   += pj * c1s[j] * t0 * __builtin_amdgcn_rcpf(bj);  // bj^(-pj-1)
  }
  if (hh == 0) priorSh[row] = prior;
  float F   = Facc[bl4];
  float gy3 = F * dpe * 2.f * enc * (1.f - enc * enc);  // dE/dy3 per row

  // g_a2 A-fragment: rank-1 (gy3 * W2) * tanh'(h2), built in-register
  v16h ga2a, ga2b;
#pragma unroll
  for (int e = 0; e < 16; ++e) {
    int K = akmap(e, hh);
    float ha = (float)h2a[e], hb = (float)h2b[e];
    ga2a[e] = (_Float16)(gy3 * W2sh[K]      * (1.f - ha * ha));
    ga2b[e] = (_Float16)(gy3 * W2sh[32 + K] * (1.f - hb * hb));
  }
  __syncthreads();
  // Backward L1: g_h1 = g_a2 @ W1^T (8 WMMAs)
#pragma unroll
  for (int t = 0; t < 4; ++t) {
    v16h bA = fw[(12 + t * 2 + 0) * 32 + lane];
    v16h bB = fw[(12 + t * 2 + 1) * 32 + lane];
    v8f c = {};
    c = wmma16(ga2a, bA, c);
    c = wmma16(ga2b, bB, c);
#pragma unroll
    for (int r = 0; r < 8; ++r)
      scr[(r + 8 * hh) * 64 + t * 16 + m] = c[r];
  }
  __syncthreads();
  v16h ga1a, ga1b;
#pragma unroll
  for (int e = 0; e < 16; ++e) {
    int K = akmap(e, hh);
    float a1 = (float)h1a[e], a2 = (float)h1b[e];
    ga1a[e] = (_Float16)(scr[m * 64 + K]      * (1.f - a1 * a1));
    ga1b[e] = (_Float16)(scr[m * 64 + 32 + K] * (1.f - a2 * a2));
  }
  // Backward L0: g_feat = g_a1 @ W0^T (2 WMMAs, single 16-col tile)
  {
    v16h bA = fw[20 * 32 + lane];
    v16h bB = fw[21 * 32 + lane];
    v8f c = {};
    c = wmma16(ga1a, bA, c);
    c = wmma16(ga1b, bB, c);
#pragma unroll
    for (int r = 0; r < 8; ++r)
      gfeatSh[(row0 + r + 8 * hh) * 16 + m] = c[r];
  }
  __syncthreads();

  // ------- Phase 5: deterministic force / g_O / S reductions --------------
  if (tid < BPB * 12) {
    int bl = tid / 12, q = tid % 12;
    const float* O = &Osh[bl * 9];
    if (q < 3) {                       // force component k
      int k = q;
      float s = 0.f;
      for (int n = 0; n < NSEL; ++n) {
        int r2 = bl * NSEL + n;
        const float* gf = &gfeatSh[r2 * 16];
        const float* NO = &NOsel[r2 * 9];
#pragma unroll
        for (int h = 0; h < 3; ++h)
          s += gf[h] * O[k * 3 + h] + gf[3 + h] * NO[k * 3 + h];
      }
      out[(long)(bb + bl) * 3 + k] = s;
    } else {                           // g_O entry (a,b)
      int a = (q - 3) / 3, b = (q - 3) % 3;
      float s = 0.f;
      for (int n = 0; n < NSEL; ++n) {
        int r2 = bl * NSEL + n;
        const float* gf = &gfeatSh[r2 * 16];
        const float* NO = &NOsel[r2 * 9];
        const float* dn = &drnSel[r2 * 3];
        float t0 = gf[b] * dn[a];                    // from dr·O term
        float t1 = 0.f;                              // from OᵀNO term
#pragma unroll
        for (int mm = 0; mm < 3; ++mm) t1 += gf[6 + 3 * b + mm] * NO[a * 3 + mm];
        s += t0 + t1;
      }
      gOAcc[bl * 9 + (q - 3)] = s;
    }
  }
  if (tid < BPB) {
    float s = 0.f;
    for (int n = 0; n < NSEL; ++n) s += priorSh[tid * NSEL + n];
    Sacc[tid] = s;
  }
  __syncthreads();

  // ------- Phase 6: energy + torque outputs -------------------------------
  if (tid < ROWS) {
    int bl = tid / NSEL, n = tid % NSEL;
    out[(long)6 * NBATCH + (long)(bb + bl) * NSEL + n] = Sacc[bl] * fcRsh[tid];
  }
  if (tid < BPB) {
    int bl = tid;
    float tq0 = 0.f, tq1 = 0.f, tq2 = 0.f;
#pragma unroll
    for (int c = 0; c < 3; ++c) {       // sum over columns c of cross(gO[:,c], O[:,c])
      float u0 = gOAcc[bl * 9 + 0 * 3 + c], u1 = gOAcc[bl * 9 + 1 * 3 + c], u2 = gOAcc[bl * 9 + 2 * 3 + c];
      float v0 = Osh[bl * 9 + 0 * 3 + c],  v1 = Osh[bl * 9 + 1 * 3 + c],  v2 = Osh[bl * 9 + 2 * 3 + c];
      tq0 += u1 * v2 - u2 * v1;
      tq1 += u2 * v0 - u0 * v2;
      tq2 += u0 * v1 - u1 * v0;
    }
    long to = (long)3 * NBATCH + (long)(bb + bl) * 3;
    out[to + 0] = tq0; out[to + 1] = tq1; out[to + 2] = tq2;
  }
}

extern "C" void kernel_launch(void* const* d_in, const int* in_sizes, int n_in,
                              void* d_out, int out_size, void* d_ws, size_t ws_size,
                              hipStream_t stream) {
  const float* dr   = (const float*)d_in[0];
  const float* ori  = (const float*)d_in[1];
  const float* nori = (const float*)d_in[2];
  const float* W0   = (const float*)d_in[3];
  const float* b0   = (const float*)d_in[4];
  const float* W1   = (const float*)d_in[5];
  const float* b1   = (const float*)d_in[6];
  const float* W2   = (const float*)d_in[7];
  const float* b2   = (const float*)d_in[8];
  const float* f1   = (const float*)d_in[9];
  const float* f2   = (const float*)d_in[10];
  float* out = (float*)d_out;
  (void)in_sizes; (void)n_in; (void)out_size; (void)d_ws; (void)ws_size;
  dim3 grid(NBATCH / BPB), block(TPB);
  energy_force_torque_fused<<<grid, block, 0, stream>>>(
      dr, ori, nori, W0, b0, W1, b1, W2, b2, f1, f2, out);
}
